// CMP_3367254360436
// MI455X (gfx1250) — matrix operand order repaired and checked by reading, not compiled, not run on hardware
//
#include <hip/hip_runtime.h>

typedef __attribute__((ext_vector_type(16))) _Float16 v16h;
typedef __attribute__((ext_vector_type(8)))  _Float16 v8h;
typedef __attribute__((ext_vector_type(8)))  float    v8f;

#define NV 2048
#define NE 8192

// ---------------- pooling: atomic f32 scatter-add over edges ----------------
// pooled layout: (V, 2, 16, 1024)  [pos block | neg block], NCHW-style
__global__ __launch_bounds__(256) void pool_kernel(
    const float* __restrict__ feats, const long long* __restrict__ edges,
    float* __restrict__ pooled)
{
    int gidx = blockIdx.x * 256 + threadIdx.x;   // NE*2*16384 threads
    int e    = gidx >> 15;
    int rem  = gidx & 32767;
    int dir  = rem >> 14;
    int elem = rem & 16383;
    long long src  = edges[e * 3 + 0];
    long long sign = edges[e * 3 + 1];
    long long dst  = edges[e * 3 + 2];
    int s = (int)(dir ? dst : src);
    int t = (int)(dir ? src : dst);
    float v = feats[s * 16384 + elem];
    int off = t * 32768 + (sign > 0 ? 0 : 16384) + elem;
    atomicAdd(&pooled[off], v);
}

// ---------------- build f16 enc_in in HWC: (V, 1024, 48) --------------------
__global__ __launch_bounds__(256) void enc_hwc_kernel(
    const float* __restrict__ feats, const float* __restrict__ pooled,
    _Float16* __restrict__ enc)
{
    int idx = blockIdx.x * 256 + threadIdx.x;    // V*1024*48 (exact multiple)
    int c  = idx % 48;
    int vp = idx / 48;              // v*1024 + p
    int v = vp >> 10, p = vp & 1023;
    float val = (c < 16) ? feats[(v << 14) + (c << 10) + p]
                         : pooled[(v << 15) + ((c - 16) << 10) + p];
    enc[idx] = (_Float16)val;
}

// ------- weight prepack: (Cout,Cin,3,3) f32 -> (9, Cout, CINP) f16 ----------
__global__ __launch_bounds__(256) void prepack_kernel(
    const float* __restrict__ w, _Float16* __restrict__ wp,
    int CIN, int CINP, int COUT)
{
    int idx = blockIdx.x * 256 + threadIdx.x;
    int total = 9 * COUT * CINP;
    if (idx >= total) return;
    int rs  = idx / (COUT * CINP);
    int rem = idx - rs * (COUT * CINP);
    int co  = rem / CINP;
    int ci  = rem - co * CINP;
    wp[idx] = (ci < CIN) ? (_Float16)w[(co * CIN + ci) * 9 + rs] : (_Float16)0.0f;
}

// ---------------- 3x3 SAME conv as 9 shifted 1x1 GEMMs via WMMA -------------
// Activations HWC; padded image (34,34,CINP) f16 in LDS; K = channels
// (contiguous) so every WMMA fragment is a wide aligned LDS load.
template <int CIN, int CINP, int COUT, typename OutT, bool HWC_OUT>
__global__ __launch_bounds__(256) void conv_kernel(
    const _Float16* __restrict__ in,   // (V, 1024, CIN)  HWC
    const _Float16* __restrict__ wrs,  // (9, COUT, CINP)
    const float*    __restrict__ bias, // (COUT)
    OutT*           __restrict__ out)
{
    constexpr int KCH  = CINP / 32;          // K-chunks per tap (2 for conv1)
    constexpr int NUMM = COUT / 16;          // 1 or 2 M-tiles
    constexpr int IMGH = 34 * 34 * CINP;     // halves
    constexpr int WH   = 9 * COUT * CINP;    // halves

    __shared__ alignas(64) _Float16 sImg[IMGH];
    __shared__ alignas(64) _Float16 sW[WH];

    const int tid = threadIdx.x;
    const int v   = blockIdx.x;

    // phase 0: zero padded image (halo + channel pad), copy weights (16B ops)
    uint4 z4; z4.x = z4.y = z4.z = z4.w = 0u;
    uint4* img4 = (uint4*)sImg;
    for (int i = tid; i < IMGH / 8; i += 256) img4[i] = z4;
    {
        const uint4* w4 = (const uint4*)wrs;
        uint4* sw4 = (uint4*)sW;
        for (int i = tid; i < WH / 8; i += 256) sw4[i] = w4[i];
    }
    __syncthreads();

    // phase 1: interior fill, pixel (y,x) -> padded (y+1,x+1), 16B copies
    {
        constexpr int QPP = CIN / 8;   // uint4 per pixel (6 or 4)
        const uint4* src4 = (const uint4*)(in + (size_t)v * 1024 * CIN);
        for (int i = tid; i < 1024 * QPP; i += 256) {
            int pix = i / QPP, q = i - pix * QPP;
            int y = pix >> 5, x = pix & 31;
            img4[(((y + 1) * 34 + (x + 1)) * CINP) / 8 + q] = src4[i];
        }
    }
    __syncthreads();

    const int wave  = tid >> 5;
    const int lane  = tid & 31;
    const int l15   = lane & 15;
    const int hi    = lane >> 4;          // wave half select
    const int nBase = wave * 8;           // 8 N-tiles per wave, both M-tiles

    // per-lane byte base address of this lane's B column (tap offset -> imm)
    int posB[8];
#pragma unroll
    for (int i = 0; i < 8; ++i) {
        int p = (nBase + i) * 16 + l15;
        int pixBase = (p >> 5) * 34 + (p & 31);          // padded (y+r, x+s) later
        posB[i] = (pixBase * CINP + hi * 16) * 2;        // bytes; 32B aligned
    }
    // per-lane byte base of A rows (one per M-tile)
    int aB[NUMM];
#pragma unroll
    for (int mt = 0; mt < NUMM; ++mt)
        aB[mt] = ((mt * 16 + l15) * CINP) * 2 + hi * 32; // bytes; 16B aligned

    const v8f vzero = {0.f, 0.f, 0.f, 0.f, 0.f, 0.f, 0.f, 0.f};
    v8f acc[NUMM][8];
#pragma unroll
    for (int mt = 0; mt < NUMM; ++mt)
#pragma unroll
        for (int i = 0; i < 8; ++i) acc[mt][i] = vzero;

#pragma unroll
    for (int rs = 0; rs < 9; ++rs) {
        const int r = rs / 3, s = rs - (rs / 3) * 3;
#pragma unroll
        for (int kc = 0; kc < KCH; ++kc) {
            // compile-time tap/chunk byte offsets -> DS immediate offsets
            const int bOfs = ((r * 34 + s) * CINP + kc * 32) * 2;
            const int aOfs = (rs * COUT * CINP + kc * 32) * 2;
            v16h a[NUMM];
#pragma unroll
            for (int mt = 0; mt < NUMM; ++mt) {
                const char* ab = (const char*)sW + aB[mt] + aOfs;
                v8h a0 = *(const v8h*)(ab);        // K = aSel+0..7
                v8h a1 = *(const v8h*)(ab + 16);   // K = aSel+8..15
                a[mt] = __builtin_shufflevector(a0, a1,
                        0, 1, 2, 3, 4, 5, 6, 7, 8, 9, 10, 11, 12, 13, 14, 15);
            }
#pragma unroll
            for (int i = 0; i < 8; ++i) {
                v16h b = *(const v16h*)((const char*)sImg + posB[i] + bOfs);
#pragma unroll
                for (int mt = 0; mt < NUMM; ++mt)
                    acc[mt][i] = __builtin_amdgcn_wmma_f32_16x16x32_f16(
                        false, a[mt], false, b, (short)0, acc[mt][i], false, false);
            }
        }
    }

    // epilogue: bias + leaky-ReLU
    // D layout: VGPR r -> M = r + (hi?8:0), N = lane&15
#pragma unroll
    for (int mt = 0; mt < NUMM; ++mt) {
        const int coBase = mt * 16 + hi * 8;
        float bv[8];
#pragma unroll
        for (int r = 0; r < 8; ++r) bv[r] = bias[coBase + r];
#pragma unroll
        for (int i = 0; i < 8; ++i) {
            int p = (nBase + i) * 16 + l15;
            float vals[8];
#pragma unroll
            for (int r = 0; r < 8; ++r) {
                float t = acc[mt][i][r] + bv[r];
                vals[r] = t > 0.f ? t : 0.1f * t;
            }
            if constexpr (HWC_OUT) {
                v8h pk;   // 8 consecutive channels -> one 16B store
#pragma unroll
                for (int r = 0; r < 8; ++r) pk[r] = (_Float16)vals[r];
                *(v8h*)((char*)out + ((size_t)(v * 1024 + p) * COUT + coBase) * 2) = pk;
            } else {
#pragma unroll
                for (int r = 0; r < 8; ++r)
                    out[((size_t)v * COUT + coBase + r) * 1024 + p] = (OutT)vals[r];
            }
        }
    }
}

extern "C" void kernel_launch(void* const* d_in, const int* in_sizes, int n_in,
                              void* d_out, int out_size, void* d_ws, size_t ws_size,
                              hipStream_t stream)
{
    const float*     feats = (const float*)d_in[0];
    const long long* edges = (const long long*)d_in[1];
    const float* W1 = (const float*)d_in[2];
    const float* b1 = (const float*)d_in[3];
    const float* W2 = (const float*)d_in[4];
    const float* b2 = (const float*)d_in[5];
    const float* W3 = (const float*)d_in[6];
    const float* b3 = (const float*)d_in[7];
    float* out = (float*)d_out;

    char* ws = (char*)d_ws;
    const size_t pooledBytes = (size_t)NV * 32768 * sizeof(float);   // 256 MiB
    float*    pooled = (float*)ws;
    _Float16* enc    = (_Float16*)(ws + pooledBytes);                // 192 MiB, HWC
    const size_t encBytes = (size_t)NV * 1024 * 48 * sizeof(_Float16);
    _Float16* w1p = (_Float16*)(ws + pooledBytes + encBytes);        // (9,32,64)
    _Float16* w2p = w1p + 9 * 32 * 64;                               // (9,32,32)
    _Float16* w3p = w2p + 9 * 32 * 32;                               // (9,16,32)
    // pooled region is dead once enc is built -> reuse for h1/h2 (exact fit)
    _Float16* h1 = (_Float16*)ws;                  // (V,1024,32) HWC, 128 MiB
    _Float16* h2 = h1 + (size_t)NV * 1024 * 32;    // (V,1024,32) HWC, 128 MiB

    hipMemsetAsync(pooled, 0, pooledBytes, stream);

    prepack_kernel<<<(9 * 32 * 64 + 255) / 256, 256, 0, stream>>>(W1, w1p, 48, 64, 32);
    prepack_kernel<<<(9 * 32 * 32 + 255) / 256, 256, 0, stream>>>(W2, w2p, 32, 32, 32);
    prepack_kernel<<<(9 * 16 * 32 + 255) / 256, 256, 0, stream>>>(W3, w3p, 32, 32, 16);

    pool_kernel<<<(NE * 2 * 16384) / 256, 256, 0, stream>>>(feats, edges, pooled);
    enc_hwc_kernel<<<((size_t)NV * 1024 * 48) / 256, 256, 0, stream>>>(feats, pooled, enc);

    conv_kernel<48, 64, 32, _Float16, true ><<<NV, 256, 0, stream>>>(enc, w1p, b1, h1);
    conv_kernel<32, 32, 32, _Float16, true ><<<NV, 256, 0, stream>>>(h1, w2p, b2, h2);
    conv_kernel<32, 32, 16, float,    false><<<NV, 256, 0, stream>>>(h2, w3p, b3, out);
}
